// StatusUpdate_11553462026852
// MI455X (gfx1250) — matrix operand order repaired and checked by reading, not compiled, not run on hardware
//
#include <hip/hip_runtime.h>
#include <hip/hip_bf16.h>

// Problem sizes (fixed by the reference)
#define B_  512
#define T_  128
#define D_  1024

// GEMM tiling: 256 threads = 8 waves; block tile 128(M) x 64(N); K-chunk = 32
#define BM  128
#define BN  64
#define KC  32
#define NC  (D_ / KC)   // 32 K-chunks in the recurrence GEMM
#define LDA 40          // padded LDS row length in bf16 elems (80 B; 16B-aligned chunks)
#define LDB 40

typedef __attribute__((ext_vector_type(16))) __bf16 v16bf;
typedef __attribute__((ext_vector_type(8)))  __bf16 v8bf;
typedef __attribute__((ext_vector_type(8)))  float  v8f;
typedef __attribute__((ext_vector_type(4)))  float  f32x4;

union frag_u { v16bf v; v8bf h[2]; };

// LDS byte offset of a shared-memory pointer (low 32 bits of the LDS aperture
// address are the offset within the wave's LDS allocation).
__device__ __forceinline__ unsigned lds_off(const void* p) {
  return (unsigned)(size_t)p;
}

// GLOBAL_LOAD_ASYNC_TO_LDS_B128: 16B per lane, memory -> LDS, ASYNCcnt-tracked.
__device__ __forceinline__ void async_copy_b128(unsigned lds, unsigned long long gaddr) {
  asm volatile("global_load_async_to_lds_b128 %0, %1, off"
               :: "v"(lds), "v"(gaddr) : "memory");
}

__device__ __forceinline__ void wait_asynccnt0() {
  asm volatile("s_wait_asynccnt 0x0" ::: "memory");
}

// ---------------------------------------------------------------------------
// One-time convert of W2 (f32 [D,D], row-major, y = x @ W2^T) to bf16.
// ---------------------------------------------------------------------------
__global__ void cvt_w2_bf16(const float* __restrict__ src,
                            __bf16* __restrict__ dst, int n) {
  int i = blockIdx.x * blockDim.x + threadIdx.x;
  int stride = gridDim.x * blockDim.x;
  for (; i < n; i += stride) dst[i] = (__bf16)src[i];
}

// ---------------------------------------------------------------------------
// GEMM1: A = [conv|attn] @ W1^T + b1  (M=512, K=2048, N=1024). Runs once.
// Also emits S1 = A + b2 (state after w_0, since state0 == 0).
// ---------------------------------------------------------------------------
__global__ __launch_bounds__(256)
void gemm1_kernel(const float* __restrict__ conv,
                  const float* __restrict__ attn,
                  const float* __restrict__ W1,   // [D, 2D] row-major
                  const float* __restrict__ b1,
                  const float* __restrict__ b2,
                  float* __restrict__ Abuf,       // [B, D]
                  float* __restrict__ S1) {       // [B, D]
  __shared__ __align__(16) __bf16 As[BM * LDA];
  __shared__ __align__(16) __bf16 Bs[BN * LDB];

  const int tid  = threadIdx.x;
  const int lane = tid & 31;
  const int wid  = tid >> 5;
  const int wm   = wid & 3;          // 4 waves along M
  const int wn   = wid >> 2;         // 2 waves along N
  const int m0   = blockIdx.y * BM;
  const int n0   = blockIdx.x * BN;

  v8f acc[2][2];
  #pragma unroll
  for (int i = 0; i < 2; ++i)
    #pragma unroll
    for (int j = 0; j < 2; ++j) acc[i][j] = v8f{};

  const int arow  = tid >> 1;          // 0..127
  const int akoff = (tid & 1) * 16;    // 0 / 16
  const int brow  = tid >> 2;          // 0..63
  const int bkoff = (tid & 3) * 8;     // 0,8,16,24

  const int fm = lane & 15;
  const int hi = lane >> 4;

  for (int kb = 0; kb < 2 * D_; kb += KC) {
    { // stage A tile: concat(conv, attn), f32 -> bf16
      const float* base = (kb < D_) ? conv : attn;
      const int    kk   = (kb < D_) ? kb : (kb - D_);
      const float* sp   = base + (size_t)(m0 + arow) * D_ + kk + akoff;
      __bf16 tmp[16];
      #pragma unroll
      for (int u = 0; u < 4; ++u) {
        f32x4 s = *(const f32x4*)(sp + 4 * u);
        tmp[4*u+0] = (__bf16)s.x; tmp[4*u+1] = (__bf16)s.y;
        tmp[4*u+2] = (__bf16)s.z; tmp[4*u+3] = (__bf16)s.w;
      }
      *(v8bf*)&As[arow * LDA + akoff]     = *(v8bf*)&tmp[0];
      *(v8bf*)&As[arow * LDA + akoff + 8] = *(v8bf*)&tmp[8];
    }
    { // stage B tile: W1[n][k], f32 -> bf16
      const float* bp = W1 + (size_t)(n0 + brow) * (2 * D_) + kb + bkoff;
      f32x4 x0 = *(const f32x4*)(bp + 0);
      f32x4 x1 = *(const f32x4*)(bp + 4);
      __bf16 tmp[8] = {(__bf16)x0.x, (__bf16)x0.y, (__bf16)x0.z, (__bf16)x0.w,
                       (__bf16)x1.x, (__bf16)x1.y, (__bf16)x1.z, (__bf16)x1.w};
      *(v8bf*)&Bs[brow * LDB + bkoff] = *(v8bf*)&tmp[0];
    }
    __syncthreads();

    frag_u af[2], bf[2];
    #pragma unroll
    for (int mi = 0; mi < 2; ++mi) {
      const __bf16* ap = &As[(wm * 32 + mi * 16 + fm) * LDA];
      af[mi].h[0] = *(const v8bf*)(ap + hi * 8);
      af[mi].h[1] = *(const v8bf*)(ap + 16 + hi * 8);
    }
    #pragma unroll
    for (int ni = 0; ni < 2; ++ni) {
      const __bf16* bp = &Bs[(wn * 32 + ni * 16 + fm) * LDB];
      bf[ni].h[0] = *(const v8bf*)(bp + hi * 8);
      bf[ni].h[1] = *(const v8bf*)(bp + 16 + hi * 8);
    }
    #pragma unroll
    for (int mi = 0; mi < 2; ++mi)
      #pragma unroll
      for (int ni = 0; ni < 2; ++ni)
        acc[mi][ni] = __builtin_amdgcn_wmma_f32_16x16x32_bf16(
            false, af[mi].v, false, bf[ni].v, (short)0, acc[mi][ni], false, false);
    __syncthreads();
  }

  #pragma unroll
  for (int mi = 0; mi < 2; ++mi)
    #pragma unroll
    for (int ni = 0; ni < 2; ++ni) {
      const int n = n0 + wn * 32 + ni * 16 + fm;
      const float bb1 = b1[n], bb2 = b2[n];
      #pragma unroll
      for (int v = 0; v < 8; ++v) {
        const int m = m0 + wm * 32 + mi * 16 + v + 8 * hi;
        const float a = acc[mi][ni][v] + bb1;
        Abuf[(size_t)m * D_ + n] = a;
        S1[(size_t)m * D_ + n]   = a + bb2;
      }
    }
}

// ---------------------------------------------------------------------------
// One recurrence step: Sout = Abuf + (Sin * w_t) @ W2^T + b2
// Double-buffered LDS; W2 tile via async global->LDS DMA; one barrier/chunk.
// ---------------------------------------------------------------------------
__global__ __launch_bounds__(256)
void step_kernel(const float* __restrict__ Sin,
                 const float* __restrict__ aw,     // [B, T, D]
                 int t,
                 const __bf16* __restrict__ W2b,   // [D, D] bf16
                 const float* __restrict__ Abuf,
                 const float* __restrict__ b2,
                 float* __restrict__ Sout) {
  __shared__ __align__(16) __bf16 As[2][BM * LDA];
  __shared__ __align__(16) __bf16 Bs[2][BN * LDB];

  const int tid  = threadIdx.x;
  const int lane = tid & 31;
  const int wid  = tid >> 5;
  const int wm   = wid & 3;
  const int wn   = wid >> 2;
  const int m0   = blockIdx.y * BM;
  const int n0   = blockIdx.x * BN;

  v8f acc[2][2];
  #pragma unroll
  for (int i = 0; i < 2; ++i)
    #pragma unroll
    for (int j = 0; j < 2; ++j) acc[i][j] = v8f{};

  const int arow  = tid >> 1;          // 0..127
  const int akoff = (tid & 1) * 16;    // 0 / 16
  const int brow  = tid >> 2;          // 0..63
  const int bkoff = (tid & 3) * 8;     // 0,8,16,24

  const int fm = lane & 15;
  const int hi = lane >> 4;

  // Per-thread stream bases (advance by KC elements per chunk)
  const float* srow = Sin + (size_t)(m0 + arow) * D_ + akoff;
  const float* wrow = aw + ((size_t)(m0 + arow) * T_ + t) * D_ + akoff;
  const unsigned long long bsrc =
      (unsigned long long)(size_t)(W2b + (size_t)(n0 + brow) * D_ + bkoff);

  // ---- prologue: stage chunk 0 into buffer 0 ----
  {
    __bf16 tmp[16];
    #pragma unroll
    for (int u = 0; u < 4; ++u) {
      f32x4 s = *(const f32x4*)(srow + 4 * u);
      f32x4 w = *(const f32x4*)(wrow + 4 * u);
      tmp[4*u+0] = (__bf16)(s.x * w.x); tmp[4*u+1] = (__bf16)(s.y * w.y);
      tmp[4*u+2] = (__bf16)(s.z * w.z); tmp[4*u+3] = (__bf16)(s.w * w.w);
    }
    *(v8bf*)&As[0][arow * LDA + akoff]     = *(v8bf*)&tmp[0];
    *(v8bf*)&As[0][arow * LDA + akoff + 8] = *(v8bf*)&tmp[8];
  }
  async_copy_b128(lds_off(&Bs[0][brow * LDB + bkoff]), bsrc);
  wait_asynccnt0();
  __syncthreads();

  for (int c = 0; c < NC; ++c) {
    const int p = c & 1, q = p ^ 1;
    const int kn = (c + 1) * KC;           // next chunk's k base
    const bool have_next = (c + 1 < NC);   // uniform: no divergence

    // ---- kick off next chunk's traffic before computing ----
    f32x4 sv[4], wv[4];
    if (have_next) {
      #pragma unroll
      for (int u = 0; u < 4; ++u) {
        sv[u] = *(const f32x4*)(srow + kn + 4 * u);
        wv[u] = *(const f32x4*)(wrow + kn + 4 * u);
      }
      async_copy_b128(lds_off(&Bs[q][brow * LDB + bkoff]),
                      bsrc + (unsigned long long)kn * sizeof(__bf16));
      if (c + 2 < NC) {  // prefetch the HBM stream one more chunk ahead
        __builtin_prefetch(wrow + kn + KC, 0, 3);
      }
    }

    // ---- compute chunk c from buffer p ----
    frag_u af[2], bf[2];
    #pragma unroll
    for (int mi = 0; mi < 2; ++mi) {
      const __bf16* ap = &As[p][(wm * 32 + mi * 16 + fm) * LDA];
      af[mi].h[0] = *(const v8bf*)(ap + hi * 8);
      af[mi].h[1] = *(const v8bf*)(ap + 16 + hi * 8);
    }
    #pragma unroll
    for (int ni = 0; ni < 2; ++ni) {
      const __bf16* bp = &Bs[p][(wn * 32 + ni * 16 + fm) * LDB];
      bf[ni].h[0] = *(const v8bf*)(bp + hi * 8);
      bf[ni].h[1] = *(const v8bf*)(bp + 16 + hi * 8);
    }
    #pragma unroll
    for (int mi = 0; mi < 2; ++mi)
      #pragma unroll
      for (int ni = 0; ni < 2; ++ni)
        acc[mi][ni] = __builtin_amdgcn_wmma_f32_16x16x32_bf16(
            false, af[mi].v, false, bf[ni].v, (short)0, acc[mi][ni], false, false);

    // ---- finish staging next A tile into buffer q ----
    if (have_next) {
      __bf16 tmp[16];
      #pragma unroll
      for (int u = 0; u < 4; ++u) {
        tmp[4*u+0] = (__bf16)(sv[u].x * wv[u].x);
        tmp[4*u+1] = (__bf16)(sv[u].y * wv[u].y);
        tmp[4*u+2] = (__bf16)(sv[u].z * wv[u].z);
        tmp[4*u+3] = (__bf16)(sv[u].w * wv[u].w);
      }
      *(v8bf*)&As[q][arow * LDA + akoff]     = *(v8bf*)&tmp[0];
      *(v8bf*)&As[q][arow * LDA + akoff + 8] = *(v8bf*)&tmp[8];
      wait_asynccnt0();   // async B tile for chunk c+1 landed in LDS
    }
    __syncthreads();       // implicit s_wait_dscnt 0 covers the ds_stores
  }

  #pragma unroll
  for (int mi = 0; mi < 2; ++mi)
    #pragma unroll
    for (int ni = 0; ni < 2; ++ni) {
      const int n = n0 + wn * 32 + ni * 16 + fm;
      const float bias = b2[n];
      #pragma unroll
      for (int v = 0; v < 8; ++v) {
        const int m = m0 + wm * 32 + mi * 16 + v + 8 * hi;
        Sout[(size_t)m * D_ + n] = Abuf[(size_t)m * D_ + n] + bias + acc[mi][ni][v];
      }
    }
}

// ---------------------------------------------------------------------------
extern "C" void kernel_launch(void* const* d_in, const int* in_sizes, int n_in,
                              void* d_out, int out_size, void* d_ws, size_t ws_size,
                              hipStream_t stream) {
  (void)in_sizes; (void)n_in; (void)out_size; (void)ws_size;
  const float* conv = (const float*)d_in[0];
  const float* attn = (const float*)d_in[1];
  const float* aw   = (const float*)d_in[2];   // [B, T, D]
  const float* W1   = (const float*)d_in[3];   // [D, 2D]
  const float* b1   = (const float*)d_in[4];
  const float* W2   = (const float*)d_in[5];   // [D, D]
  const float* b2   = (const float*)d_in[6];
  float* out = (float*)d_out;

  char* ws = (char*)d_ws;
  __bf16* W2b  = (__bf16*)(ws);                       // 2 MB
  float*  Abuf = (float*)(ws + (size_t)(2u << 20));   // 2 MB
  float*  Sa   = (float*)(ws + (size_t)(4u << 20));   // 2 MB
  float*  Sb   = (float*)(ws + (size_t)(6u << 20));   // 2 MB

  // W2 -> bf16 once per call
  cvt_w2_bf16<<<512, 256, 0, stream>>>(W2, W2b, D_ * D_);

  // A = [conv|attn] @ W1^T + b1 ; S1 = A + b2   (state after w_0, state0==0)
  const dim3 grid(D_ / BN, B_ / BM);   // (16, 4)
  gemm1_kernel<<<grid, 256, 0, stream>>>(conv, attn, W1, b1, b2, Abuf, Sa);

  // 127 sequential steps (t = 1..127), ping-pong; final writes d_out
  float* cur = Sa;
  float* nxt = Sb;
  for (int t = 1; t < T_; ++t) {
    float* dst = (t == T_ - 1) ? out : nxt;
    step_kernel<<<grid, 256, 0, stream>>>(cur, aw, t, W2b, Abuf, b2, dst);
    float* tmp = cur; cur = nxt; nxt = tmp;
  }
}
